// DAGLSTM_73366631350325
// MI455X (gfx1250) — matrix operand order repaired and checked by reading, not compiled, not run on hardware
//
#include <hip/hip_runtime.h>
#include <math.h>

// ---------------------------------------------------------------------------
// DAG-LSTM for MI455X (gfx1250): bf16 WMMA GEMM per node step, fp32 state.
//
// B=2048, N=128, IN=256, H=256, P=4.  gates = [x|h_in] @ [W_ih|W_hh]^T + bias
// Per wave: 32 rows x 16 h-cols, all 4 gates => 8 C-tiles (16x16 f32), so the
// LSTM epilogue is lane-local.  K: two 256-wide phases (x then h_in), 32 per
// v_wmma_f32_16x16x32_bf16.  Branch-free exp-based tanh/sigmoid epilogue.
// ---------------------------------------------------------------------------

typedef __attribute__((ext_vector_type(16))) __bf16 v16bf;
typedef __attribute__((ext_vector_type(8)))  __bf16 v8bf;
typedef __attribute__((ext_vector_type(8)))  float  v8f;

union ABFrag { v16bf v; v8bf h[2]; };

constexpr int B_  = 2048;
constexpr int N_  = 128;
constexpr int IN_ = 256;
constexpr int H_  = 256;
constexpr int K2  = IN_ + H_;   // 512 combined K
constexpr int G4  = 4 * H_;     // 1024 gate columns

__device__ __forceinline__ float sigf(float x) { return 1.0f / (1.0f + __expf(-x)); }

// Branch-free tanh: t = e^{-2|x|};  tanh(x) = sign(x) * (1-t)/(1+t).
// Saturates cleanly for large |x| (t -> 0), no libm branch ladder.
__device__ __forceinline__ float tanhfast(float x) {
  float t = __expf(-2.0f * __builtin_fabsf(x));
  float r = (1.0f - t) / (1.0f + t);
  return __builtin_copysignf(r, x);
}

// ---- prep: dags fp32 -> bf16 ------------------------------------------------
__global__ void k_cvt_dags(const float* __restrict__ src, __bf16* __restrict__ dst, long n) {
  long i = (long)blockIdx.x * blockDim.x + threadIdx.x;
  long stride = (long)gridDim.x * blockDim.x;
  for (; i < n; i += stride) dst[i] = (__bf16)src[i];
}

// ---- prep: pack [W_ih | W_hh] bf16 (K-contiguous rows), fold bias ----------
__global__ void k_prep_w(const float* __restrict__ w_ih, const float* __restrict__ w_hh,
                         const float* __restrict__ b_ih, const float* __restrict__ b_hh,
                         __bf16* __restrict__ wb, float* __restrict__ biasv) {
  int i = blockIdx.x * blockDim.x + threadIdx.x;
  if (i < G4 * K2) {
    int g = i / K2, k = i % K2;
    float v = (k < IN_) ? w_ih[g * IN_ + k] : w_hh[g * H_ + (k - IN_)];
    wb[i] = (__bf16)v;
  }
  if (i < G4) biasv[i] = b_ih[i] + b_hh[i];
}

// ---- prep: slot 0 of state buffers = h0/c0 ---------------------------------
__global__ void k_init_state(const float* __restrict__ h0, const float* __restrict__ c0,
                             float* __restrict__ h_buf, float* __restrict__ c_buf) {
  int i = blockIdx.x * blockDim.x + threadIdx.x;
  if (i < B_ * H_) { h_buf[i] = h0[i]; c_buf[i] = c0[i]; }
}

// ---- per step: predecessor gather-mean -> h_in (bf16) / c_in (fp32) --------
__global__ void k_gather(const float* __restrict__ h_buf, const float* __restrict__ c_buf,
                         const int* __restrict__ preds, const int* __restrict__ cnts,
                         int node, __bf16* __restrict__ hin16, float* __restrict__ cin) {
  int i = blockIdx.x * blockDim.x + threadIdx.x;    // over B*H
  if (i >= B_ * H_) return;
  int cnt = cnts[node];                             // 1..4, uniform per block
  float hs = 0.0f, cs = 0.0f;
  for (int j = 0; j < cnt; ++j) {
    long off = (long)preds[node * 4 + j] * (long)(B_ * H_) + i;
    hs += h_buf[off];
    cs += c_buf[off];
  }
  float inv = 1.0f / (float)cnt;
  hin16[i] = (__bf16)(hs * inv);
  cin[i]   = cs * inv;
}

// ---- per step: fused GEMM (bf16 WMMA) + LSTM cell --------------------------
// grid = (16, 8): block covers 128 batch rows x 32 h-cols.
// 8 waves: rowg = wave&3 (32 rows each), hcg = wave>>2 (16 h-cols each).
__global__ void __launch_bounds__(256)
k_step(const __bf16* __restrict__ dags16, const __bf16* __restrict__ hin16,
       const float* __restrict__ cin, const __bf16* __restrict__ wb,
       const float* __restrict__ biasv, int node,
       float* __restrict__ h_buf, float* __restrict__ c_buf,
       float* __restrict__ out) {
  const int lane = threadIdx.x & 31;
  const int wave = threadIdx.x >> 5;
  const int rowg = wave & 3, hcg = wave >> 2;
  const int half = lane >> 4, ln = lane & 15;
  const int r0  = blockIdx.x * 128 + rowg * 32;
  const int hc0 = blockIdx.y * 32 + hcg * 16;
  const int ncol = hc0 + ln;                 // this lane's h-column (C layout: lane=N)

  // init accumulators with bias (broadcast down the column)
  v8f acc[2][4];
  for (int q = 0; q < 4; ++q) {
    float bv = biasv[q * H_ + ncol];
    v8f a;
    for (int j = 0; j < 8; ++j) a[j] = bv;
    acc[0][q] = a; acc[1][q] = a;
  }

  // Hoisted per-row base pointers for the two K phases.
  const __bf16* xrow[2];
  const __bf16* hrow[2];
  for (int mt = 0; mt < 2; ++mt) {
    const int m = r0 + mt * 16 + ln;
    xrow[mt] = dags16 + ((long)m * N_ + node) * IN_;
    hrow[mt] = hin16 + (long)m * H_;
  }
  const __bf16* wcol[4];
  for (int q = 0; q < 4; ++q)
    wcol[q] = wb + (long)(q * H_ + hc0 + ln) * K2 + half * 16;

  // Two K phases of 256 (x @ W_ih^T, then h_in @ W_hh^T), 32 per WMMA.
  for (int phase = 0; phase < 2; ++phase) {
    const int kwoff = phase * IN_;   // offset into packed weight rows
#pragma unroll 2
    for (int kk = 0; kk < IN_; kk += 32) {
      // A fragments: lane holds row M=ln; K = kk+half*16+{0..15}.
      ABFrag afr[2];
      for (int mt = 0; mt < 2; ++mt) {
        const __bf16* rowp = phase ? hrow[mt] : xrow[mt];
        afr[mt].h[0] = *(const v8bf*)(rowp + kk + half * 16);
        afr[mt].h[1] = *(const v8bf*)(rowp + kk + half * 16 + 8);
      }
      // B fragments: lane holds column g, K = kwoff+kk+half*16 .. +15.
      v16bf bfr[4];
      for (int q = 0; q < 4; ++q)
        bfr[q] = *(const v16bf*)(wcol[q] + kwoff + kk);
      for (int mt = 0; mt < 2; ++mt)
        for (int q = 0; q < 4; ++q)
          acc[mt][q] = __builtin_amdgcn_wmma_f32_16x16x32_bf16(
              false, afr[mt].v, false, bfr[q], (short)0, acc[mt][q], false, false);
    }
  }

  // LSTM epilogue: lane-local i/f/g/o per (row, h-col). C: VGPR j -> M=j+8*half.
  const long slot = (long)(node + 1) * (long)(B_ * H_);
  for (int mt = 0; mt < 2; ++mt) {
    for (int j = 0; j < 8; ++j) {
      const int m = r0 + mt * 16 + half * 8 + j;
      const long idx = (long)m * H_ + ncol;
      const float ig = acc[mt][0][j], fg = acc[mt][1][j];
      const float gg = acc[mt][2][j], og = acc[mt][3][j];
      const float ci = cin[idx];
      const float cn = sigf(fg) * ci + sigf(ig) * tanhfast(gg);
      const float hn = sigf(og) * tanhfast(cn);
      c_buf[slot + idx] = cn;
      h_buf[slot + idx] = hn;
      if (out) out[idx] = hn;
    }
  }
}

// ---------------------------------------------------------------------------
extern "C" void kernel_launch(void* const* d_in, const int* in_sizes, int n_in,
                              void* d_out, int out_size, void* d_ws, size_t ws_size,
                              hipStream_t stream) {
  const float* dags = (const float*)d_in[0];
  const float* h0   = (const float*)d_in[1];
  const float* c0   = (const float*)d_in[2];
  const float* w_ih = (const float*)d_in[3];
  const float* w_hh = (const float*)d_in[4];
  const float* b_ih = (const float*)d_in[5];
  const float* b_hh = (const float*)d_in[6];
  const int*   preds = (const int*)d_in[7];
  const int*   cnts  = (const int*)d_in[8];
  (void)in_sizes; (void)n_in; (void)out_size; (void)ws_size;

  char* ws = (char*)d_ws;
  size_t off = 0;
  auto carve = [&](size_t bytes) {
    char* p = ws + off;
    off += (bytes + 255) & ~(size_t)255;
    return p;
  };
  const size_t stateElems = (size_t)(N_ + 1) * B_ * H_;
  float*  h_buf  = (float*)carve(stateElems * sizeof(float));        // 270.5 MB
  float*  c_buf  = (float*)carve(stateElems * sizeof(float));        // 270.5 MB
  __bf16* dags16 = (__bf16*)carve((size_t)B_ * N_ * IN_ * 2);        // 134 MB
  __bf16* wb     = (__bf16*)carve((size_t)G4 * K2 * 2);              // 1 MB
  float*  biasv  = (float*)carve((size_t)G4 * sizeof(float));
  __bf16* hin16  = (__bf16*)carve((size_t)B_ * H_ * 2);              // 1 MB
  float*  cin    = (float*)carve((size_t)B_ * H_ * sizeof(float));   // 2 MB

  const long nDag = (long)B_ * N_ * IN_;
  k_cvt_dags<<<2048, 256, 0, stream>>>(dags, dags16, nDag);
  k_prep_w<<<(G4 * K2 + 255) / 256, 256, 0, stream>>>(w_ih, w_hh, b_ih, b_hh, wb, biasv);
  k_init_state<<<(B_ * H_ + 255) / 256, 256, 0, stream>>>(h0, c0, h_buf, c_buf);

  for (int t = 0; t < N_; ++t) {
    k_gather<<<(B_ * H_ + 255) / 256, 256, 0, stream>>>(h_buf, c_buf, preds, cnts, t, hin16, cin);
    k_step<<<dim3(16, 8), 256, 0, stream>>>(dags16, hin16, cin, wb, biasv, t,
                                            h_buf, c_buf,
                                            (t == N_ - 1) ? (float*)d_out : nullptr);
  }
}
